// MFDKT_27831388078157
// MI455X (gfx1250) — matrix operand order repaired and checked by reading, not compiled, not run on hardware
//
#include <hip/hip_runtime.h>

typedef unsigned int u32;
typedef unsigned long long u64;
typedef unsigned short u16;
typedef __attribute__((ext_vector_type(16))) __bf16 v16bf;
typedef __attribute__((ext_vector_type(8)))  float  v8f;
typedef __attribute__((ext_vector_type(4)))  u32    u32x4;
typedef __attribute__((ext_vector_type(8)))  int    i32x8;
typedef __attribute__((ext_vector_type(4)))  int    i32x4;

#define B_    512
#define S_    200
#define D_    512
#define H_    512
#define G4H   2048
#define VOCAB 2002
#define BT    32            // batch rows per workgroup
#define NWG   (B_ / BT)     // 16 workgroups
#define KTOT  1024          // contraction dim = concat(x:512, h:512)
#define KT    (KTOT / 32)   // 32 k-tiles of 32
#define XSTR  520           // u16 stride per LDS row (1040 B): 4-bank shift/row, 16B aligned
#define CSTR  516           // padded fp32 stride for c state

__device__ __forceinline__ u16 f2bf(float f) {
  u32 u = __float_as_uint(f);
  u32 r = (u + 0x7FFFu + ((u >> 16) & 1u)) >> 16;   // round-to-nearest-even
  return (u16)r;
}
__device__ __forceinline__ float bf2f(u16 h) { return __uint_as_float(((u32)h) << 16); }
__device__ __forceinline__ float sigm(float x) { return 1.f / (1.f + __expf(-x)); }

// ------------------- MF extended-input gate: E[b*S+s] = sigmoid(P.Q + Pb + Qb) -------------------
__global__ void mf_gate_kernel(const int* __restrict__ user_seq, const int* __restrict__ skill_seq,
                               const float* __restrict__ P, const float* __restrict__ Q,
                               const float* __restrict__ Pb, const float* __restrict__ Qb,
                               float* __restrict__ E) {
  const int ln = threadIdx.x & 31;
  const int e  = blockIdx.x * 8 + (threadIdx.x >> 5);   // one wave per (b,s); exact fit
  const int u  = user_seq[e];
  const int sk = skill_seq[e];
  const float* pr = P + (size_t)u  * H_;
  const float* qr = Q + (size_t)sk * H_;
  float s = 0.f;
#pragma unroll
  for (int j = 0; j < 16; ++j) { int k = j * 32 + ln; s += pr[k] * qr[k]; }
#pragma unroll
  for (int off = 16; off > 0; off >>= 1) s += __shfl_xor(s, off, 32);
  if (ln == 0) E[e] = sigm(s + Pb[u] + Qb[sk]);
}

// ------------------- enc_emb fp32 -> bf16 table (TDM copies it verbatim) -------------------
__global__ void emb_bf16_kernel(const float* __restrict__ enc_emb, u16* __restrict__ Ebf) {
  const int idx = blockIdx.x * blockDim.x + threadIdx.x;   // one fp32 pair per thread
  const float2 v = ((const float2*)enc_emb)[idx];
  ((u32*)Ebf)[idx] = (u32)f2bf(v.x) | ((u32)f2bf(v.y) << 16);
}

// ------------------- Pack [W_ih | W_hh] (fp32) into bf16 WMMA B-fragment order -------------------
// Fragment f = nt*KT + kt is a 32(K)x16(N) tile: lane ln holds N = nt*16+(ln&15),
// K = kt*32 + (ln>=16 ? 16 : 0) + j, j=0..15  -> 32 contiguous bytes per lane.
__global__ void pack_w_kernel(const float* __restrict__ W_ih, const float* __restrict__ W_hh,
                              u16* __restrict__ Wpk) {
  const int t  = blockIdx.x * blockDim.x + threadIdx.x;  // one (fragment, lane) per thread
  const int f  = t >> 5;
  const int ln = t & 31;
  const int nt = f / KT;
  const int kt = f - nt * KT;
  const int n  = nt * 16 + (ln & 15);
  const int kb = kt * 32 + (ln >> 4) * 16;
  union { u16 h[16]; u32x4 q[2]; } buf;
#pragma unroll
  for (int j = 0; j < 16; ++j) {
    const int k = kb + j;
    const float w = (k < 512) ? W_ih[(size_t)n * 512 + k] : W_hh[(size_t)n * 512 + (k - 512)];
    buf.h[j] = f2bf(w);
  }
  u32x4* dst = (u32x4*)(Wpk + ((size_t)f * 512 + ln * 16));
  dst[0] = buf.q[0];
  dst[1] = buf.q[1];
}

// ------------------- Fused LSTM recurrence + decoder. One WG owns 32 batch rows for all 200 steps.
__global__ void __launch_bounds__(256, 1)
lstm_kernel(const int* __restrict__ main_input, const int* __restrict__ target_id,
            const float* __restrict__ b_ih, const float* __restrict__ b_hh,
            const float* __restrict__ dec_W, const float* __restrict__ dec_b,
            const u16* __restrict__ Wpk, const float* __restrict__ E,
            const u16* __restrict__ Ebf, float* __restrict__ out) {
  __shared__ u16   Xsh[BT][XSTR];        // x_t tile (bf16), filled by TDM gather
  __shared__ u16   Hsh[2][BT][XSTR];     // double-buffered h (bf16), A-fragment row layout
  __shared__ float Csh[BT * CSTR];       // c state, fp32
  __shared__ float bias_sh[G4H];         // b_ih + b_hh
  __shared__ float e_sh[BT];             // mf gate for current step

  const int tid    = threadIdx.x;
  const int ln     = tid & 31;
  const int w      = tid >> 5;           // 8 waves
  const int half   = ln >> 4;
  const int ncol   = ln & 15;
  const int m_base = (w & 1) * 16;       // which 16-row M tile this wave owns
  const int hgrp   = w >> 1;             // 4 wave-groups over 32 hidden-column tiles
  const int b0     = blockIdx.x * BT;
  const int arow   = m_base + ncol;      // A-fragment row for this lane

  for (int idx = tid; idx < G4H; idx += 256) bias_sh[idx] = b_ih[idx] + b_hh[idx];
  for (int idx = tid; idx < BT * H_; idx += 256) {
    const int r = idx >> 9, c = idx & 511;
    Hsh[0][r][c] = 0;                    // h0 = 0 (buffer 0)
    Csh[r * CSTR + c] = 0.f;             // c0 = 0
  }

  // --- Tensor DMA descriptor (gather mode, 16-bit row indices, 16 rows of 512 bf16) ---
  // pad_enable: insert 4 DWORDs (16B) after every 256 DWORDs (one row) -> LDS row stride 1040B.
  u32x4 g0; i32x8 g1;
  {
    const u32 lds_base = (u32)(uintptr_t)&Xsh[(w & 1) * 16][0];   // low 32 bits = LDS byte offset
    const u64 ga = (u64)(uintptr_t)Ebf;
    g0[0] = 0x80000001u;                               // count=1 | gather_mode=1 | 16-bit indices
    g0[1] = lds_base;                                  // lds_addr
    g0[2] = (u32)ga;                                   // global_addr[31:0]
    g0[3] = (u32)((ga >> 32) & 0x1FFFFFFu) | 0x80000000u;  // global_addr[56:32] | type=2
    g1[0] = (int)((1u << 16) | (1u << 20) | (7u << 22) | (3u << 25)); // data_size=2B, pad 256dw/+4dw
    g1[1] = (int)(512u << 16);                         // tensor_dim0 = 512 (lo16 in [31:16])
    g1[2] = (int)((u32)VOCAB << 16);                   // tensor_dim1 = 2002
    g1[3] = (int)(512u << 16);                         // tile_dim0 = 512
    g1[4] = 16;                                        // tile_dim1 = #valid indices = 16
    g1[5] = 512;                                       // tensor_dim0_stride = 512 elements
    g1[6] = 0;
    g1[7] = 0;
  }
  __syncthreads();

  for (int t = 0; t < S_; ++t) {
    const int p = t & 1;                 // h read-buffer for this step

    // ---- phase 1: TDM row-gather x_t = Ebf[main_input[:,t]] straight into LDS ----
    if (w < 2) {
      i32x4 g2, g3;
#pragma unroll
      for (int r = 0; r < 8; ++r) {      // pack 16 token indices (wave-uniform scalars)
        const u32 t0 = (u32)main_input[(size_t)(b0 + w * 16 + 2 * r) * S_ + t];
        const u32 t1 = (u32)main_input[(size_t)(b0 + w * 16 + 2 * r + 1) * S_ + t];
        const u32 pk = t0 | (t1 << 16);
        if (r < 4) g2[r] = (int)pk; else g3[r - 4] = (int)pk;
      }
      const i32x8 gz = {};               // 6-arg toolchain: extra group, zero-filled
      __builtin_amdgcn_tensor_load_to_lds(g0, g1, g2, g3, gz, 0);
      __builtin_amdgcn_s_wait_tensorcnt(0);
    }
    if (tid < BT) e_sh[tid] = E[(size_t)(b0 + tid) * S_ + t];
    __syncthreads();

    // ---- phase 2: gates = [x|h] @ Wcat^T + b  (bf16 WMMA, f32 acc), then cell update ----
    const u16* abase_x = &Xsh[arow][half * 16];          // loop-invariant A bases (no per-kt select)
    const u16* abase_h = &Hsh[p][arow][half * 16];

#pragma unroll 1
    for (int j = 0; j < 8; ++j) {
      const int ht  = hgrp * 8 + j;          // hidden-column tile 0..31
      const int col = ht * 16 + ncol;        // hidden column 0..511 for this lane
      v8f acc[4];                            // i, f, g, o accumulators (16x16 f32 tiles)
#pragma unroll
      for (int g = 0; g < 4; ++g) {
        const float bi = bias_sh[g * 512 + col];
#pragma unroll
        for (int i = 0; i < 8; ++i) acc[g][i] = bi;
      }

      // one 16x16x32 MMA group: A fragment from LDS, 4 gate B fragments streamed from L2
      auto mma_step = [&](const u16* aptr, int kt) {
        union { v16bf v; u32x4 q[2]; } af;
        af.q[0] = *(const u32x4*)aptr;
        af.q[1] = *(const u32x4*)(aptr + 8);
        if (kt + 1 < KT)                     // -> global_prefetch_b8
          __builtin_prefetch(Wpk + ((size_t)(ht * KT + kt + 1) * 512 + ln * 16), 0, 3);
#pragma unroll
        for (int g = 0; g < 4; ++g) {
          const int nt = g * 32 + ht;        // gate g columns start at g*512
          union { v16bf v; u32x4 q[2]; } bfr;
          const u32x4* wq = (const u32x4*)(Wpk + ((size_t)(nt * KT + kt) * 512 + ln * 16));
          bfr.q[0] = wq[0];
          bfr.q[1] = wq[1];
          acc[g] = __builtin_amdgcn_wmma_f32_16x16x32_bf16(
              false, af.v, false, bfr.v, (short)0, acc[g], false, false);
        }
      };

#pragma unroll 2
      for (int kt = 0; kt < 16; ++kt)        // x half of contraction (K 0..511)
        mma_step(abase_x + kt * 32, kt);
#pragma unroll 2
      for (int kt = 0; kt < 16; ++kt)        // h half of contraction (K 512..1023)
        mma_step(abase_h + kt * 32, 16 + kt);

      // cell update: c = sig(f)*(c*(1+e)) + sig(i)*tanh(g); h = sig(o)*tanh(c)
#pragma unroll
      for (int r = 0; r < 8; ++r) {
        const int row = m_base + r + half * 8;   // C/D layout: M = r + 8*half
        const float e = e_sh[row];
        float c = Csh[row * CSTR + col];
        c += c * e;
        const float iv = sigm(acc[0][r]);
        const float fv = sigm(acc[1][r]);
        const float gv = tanhf(acc[2][r]);
        const float ov = sigm(acc[3][r]);
        c = fv * c + iv * gv;
        Csh[row * CSTR + col] = c;
        Hsh[p ^ 1][row][col] = f2bf(ov * tanhf(c));   // write h_{t+1} to other buffer
      }
    }
    __syncthreads();
  }

  // ---- decoder: out[b] = sigmoid(h . dec_W[target] + dec_b[target]); final h in buffer 0 ----
  for (int r = 0; r < 4; ++r) {
    const int row = w * 4 + r;
    const int b   = b0 + row;
    const int tg  = target_id[b];
    const float* dw = dec_W + (size_t)tg * H_;
    float s = 0.f;
#pragma unroll
    for (int j = 0; j < 16; ++j) {
      const int k = j * 32 + ln;
      s += bf2f(Hsh[0][row][k]) * dw[k];
    }
#pragma unroll
    for (int off = 16; off > 0; off >>= 1) s += __shfl_xor(s, off, 32);
    if (ln == 0) out[b] = sigm(s + dec_b[tg]);
  }
}

extern "C" void kernel_launch(void* const* d_in, const int* in_sizes, int n_in,
                              void* d_out, int out_size, void* d_ws, size_t ws_size,
                              hipStream_t stream) {
  const int*   main_input = (const int*)d_in[0];
  const int*   target_id  = (const int*)d_in[1];
  const int*   user_seq   = (const int*)d_in[2];
  const int*   skill_seq  = (const int*)d_in[3];
  const float* enc_emb    = (const float*)d_in[4];
  const float* P          = (const float*)d_in[5];
  const float* Q          = (const float*)d_in[6];
  const float* Pb         = (const float*)d_in[7];
  const float* Qb         = (const float*)d_in[8];
  const float* W_ih       = (const float*)d_in[9];
  const float* W_hh       = (const float*)d_in[10];
  const float* b_ih       = (const float*)d_in[11];
  const float* b_hh       = (const float*)d_in[12];
  const float* dec_W      = (const float*)d_in[13];
  const float* dec_b      = (const float*)d_in[14];

  float* E   = (float*)d_ws;                                        // B*S fp32 (0.4 MB)
  u16*   Wpk = (u16*)((char*)d_ws + (size_t)B_ * S_ * 4);           // 4 MB bf16 packed weights
  u16*   Ebf = (u16*)((char*)d_ws + (size_t)B_ * S_ * 4 + (size_t)G4H * KTOT * 2); // 2 MB bf16 emb

  mf_gate_kernel<<<(B_ * S_) / 8, 256, 0, stream>>>(user_seq, skill_seq, P, Q, Pb, Qb, E);
  emb_bf16_kernel<<<(VOCAB * D_ / 2) / 256, 256, 0, stream>>>(enc_emb, Ebf);
  pack_w_kernel<<<(128 * KT * 32) / 256, 256, 0, stream>>>(W_ih, W_hh, Wpk);
  lstm_kernel<<<NWG, 256, 0, stream>>>(main_input, target_id, b_ih, b_hh,
                                       dec_W, dec_b, Wpk, E, Ebf, (float*)d_out);
}
